// PointNetSetAbstraction_17952963297794
// MI455X (gfx1250) — compile-verified, matrix operand order checked by reading, not run on hardware
//
#include <hip/hip_runtime.h>
#include <hip/hip_bf16.h>

// ---------------------------------------------------------------------------
// PointNet++ Set Abstraction for MI455X (gfx1250, wave32, WMMA)
// B=8, N=4096, D=64, NPOINT=1024, NSAMPLE=32, RADIUS=0.2, MLP 67->64->64->128
// ---------------------------------------------------------------------------

#define Bx 8
#define Nx 4096
#define Dx 64
#define NPOINT 1024
#define NSAMPLE 32
#define RADIUS2 (0.2f * 0.2f)
#define INCH 67
#define KPAD0 96   // 67 padded up to multiple of 32

typedef __attribute__((ext_vector_type(16))) _Float16 v16h;
typedef __attribute__((ext_vector_type(8)))  float    v8f;

struct alignas(16) H8 { _Float16 h[8]; };

// Load a 16x32 f16 fragment (A-layout; B uses same layout with N as "row")
// from row-major f16 LDS storage with given row stride.
__device__ inline v16h load_frag(const _Float16* base, int stride, int row0,
                                 int k0, int lane) {
  int r  = row0 + (lane & 15);
  int kb = k0 + ((lane >> 4) << 3);       // lanes 0-15: K 0..7 ; 16-31: K 8..15
  const _Float16* p = base + r * stride + kb;
  H8 lo = *reinterpret_cast<const H8*>(p);        // K kb..kb+7
  H8 hi = *reinterpret_cast<const H8*>(p + 16);   // K kb+16..kb+23
  v16h out;
#pragma unroll
  for (int i = 0; i < 8; ++i) { out[i] = lo.h[i]; out[8 + i] = hi.h[i]; }
  return out;
}

__device__ inline v8f wmma32(v16h a, v16h b, v8f c) {
  // D = A(16x32 f16) * B(32x16 f16) + C(16x16 f32)
  return __builtin_amdgcn_wmma_f32_16x16x32_f16(
      false, a, false, b, (short)0, c, false, false);
}

// ---------------------------------------------------------------------------
// Kernel 1: farthest point sampling. One block (256 threads) per batch.
// Points live in registers (16 per thread, strided); centroid broadcast via
// 3 floats of LDS; arg-max via LDS tree. Writes new_xyz directly to d_out.
// ---------------------------------------------------------------------------
__global__ __launch_bounds__(256) void fps_kernel(const float* __restrict__ xyz,
                                                  float* __restrict__ out_xyz) {
  const int b = blockIdx.x;
  const int t = threadIdx.x;
  const float* bx = xyz + (size_t)b * Nx * 3;

  float px[16], py[16], pz[16], d[16];
#pragma unroll
  for (int i = 0; i < 16; ++i) {
    int p = t + (i << 8);
    px[i] = bx[p * 3 + 0];
    py[i] = bx[p * 3 + 1];
    pz[i] = bx[p * 3 + 2];
    d[i]  = 1e10f;
  }

  __shared__ float rv[256];
  __shared__ int   ri[256];
  __shared__ float cent[3];
  __shared__ int   sfar;
  if (t == 0) sfar = 0;
  __syncthreads();

  for (int it = 0; it < NPOINT; ++it) {
    int far = sfar;
    if ((far & 255) == t) {               // owner publishes centroid coords
      int slot = far >> 8;
      cent[0] = px[slot]; cent[1] = py[slot]; cent[2] = pz[slot];
    }
    __syncthreads();
    float cx = cent[0], cy = cent[1], cz = cent[2];
    if (t == 0) {
      float* o = out_xyz + ((size_t)b * NPOINT + it) * 3;
      o[0] = cx; o[1] = cy; o[2] = cz;
    }
    float bestv = -1.f; int besti = 0;
#pragma unroll
    for (int i = 0; i < 16; ++i) {
      float dx = px[i] - cx, dy = py[i] - cy, dz = pz[i] - cz;
      float nd = fminf(d[i], dx * dx + dy * dy + dz * dz);
      d[i] = nd;
      int gi = t + (i << 8);
      if (nd > bestv || (nd == bestv && gi < besti)) { bestv = nd; besti = gi; }
    }
    rv[t] = bestv; ri[t] = besti;
    __syncthreads();
    for (int s = 128; s > 0; s >>= 1) {
      if (t < s) {
        float ov = rv[t + s]; int oi = ri[t + s];
        if (ov > rv[t] || (ov == rv[t] && oi < ri[t])) { rv[t] = ov; ri[t] = oi; }
      }
      __syncthreads();
    }
    if (t == 0) sfar = ri[0];
    __syncthreads();
  }
}

// ---------------------------------------------------------------------------
// Kernel 2: ball query. One wave32 per centroid; ballot + prefix-popcount
// produces the first 32 in-radius indices in ascending order (matches the
// reference's sort-of-masked-indices semantics), padded with the first hit.
// ---------------------------------------------------------------------------
__global__ __launch_bounds__(256) void ballq_kernel(const float* __restrict__ xyz,
                                                    const float* __restrict__ new_xyz,
                                                    int* __restrict__ gidx) {
  int wid  = (blockIdx.x << 3) + (threadIdx.x >> 5);
  int lane = threadIdx.x & 31;
  int b = wid >> 10, s = wid & (NPOINT - 1);
  const float* bx = xyz + (size_t)b * Nx * 3;
  const float* c  = new_xyz + ((size_t)b * NPOINT + s) * 3;
  float cx = c[0], cy = c[1], cz = c[2];
  int* g = gidx + ((size_t)b * NPOINT + s) * NSAMPLE;

  int count = 0, first = 0;
  for (int j = 0; j < Nx && count < NSAMPLE; j += 32) {
    int p = j + lane;
    float dx = bx[p * 3 + 0] - cx;
    float dy = bx[p * 3 + 1] - cy;
    float dz = bx[p * 3 + 2] - cz;
    bool in = (dx * dx + dy * dy + dz * dz) <= RADIUS2;
    unsigned mask = (unsigned)__ballot(in);
    if (count == 0 && mask) first = j + __ffs(mask) - 1;
    int before = __popc(mask & ((1u << lane) - 1u));
    int pos = count + before;
    if (in && pos < NSAMPLE) g[pos] = p;
    count += __popc(mask);
  }
  if (count < NSAMPLE)
    for (int p2 = count + lane; p2 < NSAMPLE; p2 += 32) g[p2] = first;
}

// ---------------------------------------------------------------------------
// Kernel 3: gather + 3-layer MLP (WMMA f16->f32) + max-pool.
// One block (256 thr = 8 waves) handles 4 centroids = 128 rows.
// All activations/weights staged in LDS as f16; BN+bias folded into per-
// channel affine (A,B); layer-2 max-pool done in registers.
// ---------------------------------------------------------------------------
__global__ __launch_bounds__(256) void mlp_kernel(
    const float* __restrict__ xyz, const float* __restrict__ points,
    const float* __restrict__ w0, const float* __restrict__ b0,
    const float* __restrict__ g0, const float* __restrict__ be0,
    const float* __restrict__ m0, const float* __restrict__ v0,
    const float* __restrict__ w1, const float* __restrict__ b1,
    const float* __restrict__ g1, const float* __restrict__ be1,
    const float* __restrict__ m1, const float* __restrict__ v1,
    const float* __restrict__ w2, const float* __restrict__ b2,
    const float* __restrict__ g2, const float* __restrict__ be2,
    const float* __restrict__ m2, const float* __restrict__ v2,
    const float* __restrict__ new_xyz, const int* __restrict__ gidx,
    float* __restrict__ out_pts) {
  __shared__ _Float16 X0[128 * KPAD0];   // layer-0 input; reused as X2
  __shared__ _Float16 X1[128 * 64];
  __shared__ _Float16 W0s[64 * KPAD0];
  __shared__ _Float16 W1s[64 * 64];
  __shared__ _Float16 W2s[128 * 64];
  __shared__ float aA0[64], aB0[64], aA1[64], aB1[64], aA2[128], aB2[128];

  const int t   = threadIdx.x;
  const int blk = blockIdx.x;        // 2048 blocks: 256 per batch x 4 centroids
  const int b   = blk >> 8;
  const int s0  = (blk & 255) << 2;
  const int w   = t >> 5;            // wave id 0..7
  const int lane = t & 31;

  // ---- stage weights (f16, zero-padded K) + folded BN affine ----
  for (int i = t; i < 64 * KPAD0; i += 256) {
    int o = i / KPAD0, c = i % KPAD0;
    W0s[i] = (c < INCH) ? (_Float16)w0[o * INCH + c] : (_Float16)0.f;
  }
  for (int i = t; i < 64 * 64; i += 256)  W1s[i] = (_Float16)w1[i];
  for (int i = t; i < 128 * 64; i += 256) W2s[i] = (_Float16)w2[i];
  if (t < 64) {
    float s = g0[t] * rsqrtf(v0[t] + 1e-5f);
    aA0[t] = s; aB0[t] = (b0[t] - m0[t]) * s + be0[t];
    s = g1[t] * rsqrtf(v1[t] + 1e-5f);
    aA1[t] = s; aB1[t] = (b1[t] - m1[t]) * s + be1[t];
  }
  if (t < 128) {
    float s = g2[t] * rsqrtf(v2[t] + 1e-5f);
    aA2[t] = s; aB2[t] = (b2[t] - m2[t]) * s + be2[t];
  }

  // ---- gather: 128 rows x 96 ch (xyz-rel | points | zero pad) ----
  {
    int row = t >> 1, half = t & 1;
    int grp = row >> 5, k = row & 31;
    int s = s0 + grp;
    int idx = gidx[(((size_t)b << 10) + s) * NSAMPLE + k];
    const float* nc   = new_xyz + (((size_t)b << 10) + s) * 3;
    const float* pxyz = xyz + ((size_t)b * Nx + idx) * 3;
    const float* ppts = points + ((size_t)b * Nx + idx) * Dx;
    int lo = half * 48, hi = lo + 48;
    for (int ch = lo; ch < hi; ++ch) {
      float val;
      if (ch < 3)         val = pxyz[ch] - nc[ch];
      else if (ch < INCH) val = ppts[ch - 3];
      else                val = 0.f;
      X0[row * KPAD0 + ch] = (_Float16)val;
    }
  }
  __syncthreads();

  // ---- layer 0: (128x96) x (96x64) -> X1, fused affine+ReLU ----
  for (int tt = w; tt < 32; tt += 8) {
    int mI = tt >> 2, nI = tt & 3;
    v8f c = {};
#pragma unroll
    for (int ks = 0; ks < 3; ++ks) {
      v16h a  = load_frag(X0,  KPAD0, mI * 16, ks * 32, lane);
      v16h bf = load_frag(W0s, KPAD0, nI * 16, ks * 32, lane);
      c = wmma32(a, bf, c);
    }
    int col = nI * 16 + (lane & 15);
    float A = aA0[col], Bc = aB0[col];
    int rbase = mI * 16 + ((lane >> 4) << 3);
#pragma unroll
    for (int vv = 0; vv < 8; ++vv) {
      float y = fmaxf(c[vv] * A + Bc, 0.f);
      X1[(rbase + vv) * 64 + col] = (_Float16)y;
    }
  }
  __syncthreads();

  // ---- layer 1: (128x64) x (64x64) -> X2 (aliases X0) ----
  _Float16* X2 = X0;
  for (int tt = w; tt < 32; tt += 8) {
    int mI = tt >> 2, nI = tt & 3;
    v8f c = {};
#pragma unroll
    for (int ks = 0; ks < 2; ++ks) {
      v16h a  = load_frag(X1,  64, mI * 16, ks * 32, lane);
      v16h bf = load_frag(W1s, 64, nI * 16, ks * 32, lane);
      c = wmma32(a, bf, c);
    }
    int col = nI * 16 + (lane & 15);
    float A = aA1[col], Bc = aB1[col];
    int rbase = mI * 16 + ((lane >> 4) << 3);
#pragma unroll
    for (int vv = 0; vv < 8; ++vv) {
      float y = fmaxf(c[vv] * A + Bc, 0.f);
      X2[(rbase + vv) * 64 + col] = (_Float16)y;
    }
  }
  __syncthreads();

  // ---- layer 2: (128x64) x (64x128) + register max-pool over 32 rows ----
  // Wave w owns N-tile w (cols w*16..w*16+15) for all 8 M-tiles.
  {
    int nt = w;
    int col = nt * 16 + (lane & 15);
    float A = aA2[col], Bc = aB2[col];
    for (int grp = 0; grp < 4; ++grp) {
      float mx = 0.f;   // ReLU floor doubles as max-identity
#pragma unroll
      for (int half2 = 0; half2 < 2; ++half2) {
        int mI = grp * 2 + half2;
        v8f c = {};
#pragma unroll
        for (int ks = 0; ks < 2; ++ks) {
          v16h a  = load_frag(X2,  64, mI * 16, ks * 32, lane);
          v16h bf = load_frag(W2s, 64, nt * 16, ks * 32, lane);
          c = wmma32(a, bf, c);
        }
#pragma unroll
        for (int vv = 0; vv < 8; ++vv) mx = fmaxf(mx, c[vv] * A + Bc);
      }
      mx = fmaxf(mx, __shfl_xor(mx, 16, 32));   // combine lane halves (rows)
      if (lane < 16)
        out_pts[(((size_t)b << 10) + s0 + grp) * 128 + col] = mx;
    }
  }
}

// ---------------------------------------------------------------------------
extern "C" void kernel_launch(void* const* d_in, const int* in_sizes, int n_in,
                              void* d_out, int out_size, void* d_ws, size_t ws_size,
                              hipStream_t stream) {
  (void)in_sizes; (void)n_in; (void)out_size; (void)ws_size;
  const float* xyz    = (const float*)d_in[0];
  const float* points = (const float*)d_in[1];
  const float* w0 = (const float*)d_in[2];
  const float* b0 = (const float*)d_in[3];
  const float* g0 = (const float*)d_in[4];
  const float* be0 = (const float*)d_in[5];
  const float* m0 = (const float*)d_in[6];
  const float* v0 = (const float*)d_in[7];
  const float* w1 = (const float*)d_in[8];
  const float* b1 = (const float*)d_in[9];
  const float* g1 = (const float*)d_in[10];
  const float* be1 = (const float*)d_in[11];
  const float* m1 = (const float*)d_in[12];
  const float* v1 = (const float*)d_in[13];
  const float* w2 = (const float*)d_in[14];
  const float* b2 = (const float*)d_in[15];
  const float* g2 = (const float*)d_in[16];
  const float* be2 = (const float*)d_in[17];
  const float* m2 = (const float*)d_in[18];
  const float* v2 = (const float*)d_in[19];

  float* out_xyz = (float*)d_out;                          // B*1024*3
  float* out_pts = out_xyz + (size_t)Bx * NPOINT * 3;      // B*1024*128
  int*   gidx    = (int*)d_ws;                             // B*1024*32 ints

  fps_kernel<<<Bx, 256, 0, stream>>>(xyz, out_xyz);
  ballq_kernel<<<(Bx * NPOINT) / 8, 256, 0, stream>>>(xyz, out_xyz, gidx);
  mlp_kernel<<<(Bx * NPOINT) / 4, 256, 0, stream>>>(
      xyz, points, w0, b0, g0, be0, m0, v0, w1, b1, g1, be1, m1, v1,
      w2, b2, g2, be2, m2, v2, out_xyz, gidx, out_pts);
}
